// GraphPlanEncoder_66829691125762
// MI455X (gfx1250) — matrix operand order repaired and checked by reading, not compiled, not run on hardware
//
#include <hip/hip_runtime.h>
#include <hip/hip_bf16.h>

#define N_NODES 50000
#define N_EDGES 800000
#define HID 128
#define OUT_DIM 256
#define N_GRAPHS 256

typedef __attribute__((ext_vector_type(16))) __bf16    v16bf;
typedef __attribute__((ext_vector_type(8)))  float     v8f;
typedef __attribute__((ext_vector_type(4)))  unsigned  v4u;

__device__ __forceinline__ unsigned short f2bf_bits(float f) {
    unsigned u = __float_as_uint(f);
    unsigned r = u + 0x7FFFu + ((u >> 16) & 1u);   // round-to-nearest-even
    return (unsigned short)(r >> 16);
}

// ---------------------------------------------------------------- zero fill
__global__ void zero_f(float* __restrict__ p, int n) {
    for (int i = blockIdx.x * blockDim.x + threadIdx.x; i < n;
         i += gridDim.x * blockDim.x)
        p[i] = 0.f;
}

// ------------------------------------------------- pack weights into WMMA B
// Wcat[k][n], k<128 -> Wn[k][n], k>=128 -> Ws[k-128][n]. For n-tile t (8),
// k-step s (8), lane L: 16 contiguous bf16 = VGPRs v=0..7, pairs p=0..1 with
// k = s*32 + (L>>4)*16 + 2v + p, n = t*16 + (L&15). (contiguous k order, same as the A fragment)
__global__ void pack_w(const float* __restrict__ Wn, const float* __restrict__ Ws,
                       unsigned short* __restrict__ out) {
    int t = blockIdx.x >> 3;
    int s = blockIdx.x & 7;
    int lane = threadIdx.x;                 // 32 threads
    int n  = t * 16 + (lane & 15);
    int kb = s * 32 + (lane >> 4) * 16;
    unsigned short* dst = out + (((t * 8 + s) * 32 + lane) << 4);
    #pragma unroll
    for (int v = 0; v < 8; ++v)
        #pragma unroll
        for (int p = 0; p < 2; ++p) {
            int k = kb + 2 * v + p;
            float val = (k < HID) ? Wn[k * HID + n] : Ws[(k - HID) * HID + n];
            dst[v * 2 + p] = f2bf_bits(val);
        }
}

// ---------------------------------------------------------- edge scatter-add
__global__ __launch_bounds__(256) void scatter_edges(
    const float* __restrict__ h, const int* __restrict__ src,
    const int* __restrict__ dst, float* __restrict__ agg,
    float* __restrict__ cntb) {
    int e = blockIdx.x * 8 + (threadIdx.x >> 5);
    if (e >= N_EDGES) return;
    int lane = threadIdx.x & 31;
    int s = src[e], d = dst[e];
    const float4 v = *(const float4*)(h + s * HID + lane * 4);
    float* o = agg + d * HID + lane * 4;
    atomicAdd(o + 0, v.x); atomicAdd(o + 1, v.y);
    atomicAdd(o + 2, v.z); atomicAdd(o + 3, v.w);
    if (lane == 0) atomicAdd(cntb + d, 1.f);
}

// --------------------------------------- SAGE block: WMMA GEMM + LN + ReLU
// hout = relu(LN( [agg/cnt | x] @ Wcat + b ))    (K = 256, N = 128)
// 256 threads = 8 waves; block owns 16 rows; wave w owns N-tile [16w,16w+16)
__global__ __launch_bounds__(256) void sage_gemm_ln_relu(
    const float* __restrict__ agg, const float* __restrict__ cnt,
    const float* __restrict__ xin, const unsigned short* __restrict__ wpack,
    const float* __restrict__ bias, const float* __restrict__ gam,
    const float* __restrict__ bet, float* __restrict__ hout) {
    __shared__ __align__(16) unsigned short ldsA[16 * 264]; // 16 rows x 256 bf16 (+pad)
    __shared__ float hbuf[16 * 128];
    __shared__ float psum[16 * 16];
    __shared__ float psq[16 * 16];
    __shared__ float mstat[16], rstat[16];

    const int tid = threadIdx.x;
    const int m0 = blockIdx.x * 16;

    // stage A = [agg/max(cnt,1) | x] as bf16 rows of 256
    for (int i = 0; i < 16; ++i) {
        int flat = i * 256 + tid;
        int row = flat >> 8;
        int col = flat & 255;
        int node = m0 + row;
        float val = 0.f;
        if (node < N_NODES) {
            if (col < HID) {
                float c = cnt[node];
                val = agg[node * HID + col] / fmaxf(c, 1.f);
            } else {
                val = xin[node * HID + (col - HID)];
            }
        }
        ldsA[row * 264 + col] = f2bf_bits(val);
    }
    __syncthreads();

    const int wv   = tid >> 5;
    const int lane = tid & 31;
    const int mm   = lane & 15;
    const int hi   = lane >> 4;
    const int nb   = wv * 16;

    v8f c = {};
    #pragma unroll
    for (int s = 0; s < 8; ++s) {
        // A fragment: k = hi*16 + e (contiguous, matches B packing): two 16B chunks
        union { v4u q[2]; v16bf v; } a;
        const unsigned short* ap = &ldsA[mm * 264 + s * 32 + hi * 16];
        a.q[0] = *(const v4u*)(ap);
        a.q[1] = *(const v4u*)(ap + 8);
        // B fragment: pre-packed, 32B contiguous per lane
        v16bf bv = *(const v16bf*)(wpack + (((wv * 8 + s) * 32 + lane) << 4));
        c = __builtin_amdgcn_wmma_f32_16x16x32_bf16(
                false, a.v, false, bv, (short)0, c, false, false);
    }

    // C layout: VGPR r -> M = hi*8 + r, N = nb + mm
    float bval = bias[nb + mm];
    #pragma unroll
    for (int r = 0; r < 8; ++r)
        hbuf[(hi * 8 + r) * 128 + nb + mm] = c[r] + bval;
    __syncthreads();

    // LayerNorm stats: 16 threads per row, 8 cols each
    {
        int row = tid >> 4, seg = tid & 15;
        float s1 = 0.f, s2 = 0.f;
        #pragma unroll
        for (int j = 0; j < 8; ++j) {
            float v = hbuf[row * 128 + seg * 8 + j];
            s1 += v; s2 += v * v;
        }
        psum[row * 16 + seg] = s1;
        psq [row * 16 + seg] = s2;
    }
    __syncthreads();
    if (tid < 16) {
        float s1 = 0.f, s2 = 0.f;
        for (int j = 0; j < 16; ++j) { s1 += psum[tid * 16 + j]; s2 += psq[tid * 16 + j]; }
        float mean = s1 * (1.f / 128.f);
        float var  = s2 * (1.f / 128.f) - mean * mean;
        mstat[tid] = mean;
        rstat[tid] = rsqrtf(var + 1e-5f);
    }
    __syncthreads();
    {
        int row = tid >> 4, seg = tid & 15;
        int node = m0 + row;
        if (node < N_NODES) {
            float mean = mstat[row], rr = rstat[row];
            #pragma unroll
            for (int j = 0; j < 8; ++j) {
                int col = seg * 8 + j;
                float v = (hbuf[row * 128 + col] - mean) * rr * gam[col] + bet[col];
                hout[node * HID + col] = fmaxf(v, 0.f);
            }
        }
    }
}

// ------------------------------------------------------------ graph pooling
__global__ void pool_nodes(const float* __restrict__ h, const int* __restrict__ gb,
                           float* __restrict__ gsum, unsigned* __restrict__ gmax,
                           float* __restrict__ gcnt) {
    int n = blockIdx.x;
    int k = threadIdx.x;                    // 128
    int g = gb[n];
    float v = h[n * HID + k];
    atomicAdd(&gsum[g * HID + k], v);
    unsigned bits = __float_as_uint(v);
    unsigned enc = bits ^ ((bits & 0x80000000u) ? 0xFFFFFFFFu : 0x80000000u);
    atomicMax(&gmax[g * HID + k], enc);
    if (k == 0) atomicAdd(&gcnt[g], 1.f);
}

__global__ void pool_finalize(const float* __restrict__ gsum,
                              const unsigned* __restrict__ gmax,
                              const float* __restrict__ gcnt,
                              float* __restrict__ pooled) {
    int g = blockIdx.x, k = threadIdx.x;    // 128
    float c = gcnt[g];
    pooled[g * 256 + k] = gsum[g * HID + k] / fmaxf(c, 1.f);
    float mx = 0.f;
    if (c > 0.f) {
        unsigned enc = gmax[g * HID + k];
        unsigned bits = (enc & 0x80000000u) ? (enc ^ 0x80000000u) : ~enc;
        mx = __uint_as_float(bits);
    }
    pooled[g * 256 + 128 + k] = mx;
}

// ----------------------------------------------------- readout + normalize
__global__ __launch_bounds__(256) void readout(
    const float* __restrict__ pooled, const float* __restrict__ W1,
    const float* __restrict__ b1, const float* __restrict__ W2,
    const float* __restrict__ b2, float* __restrict__ out) {
    __shared__ float pr[256];
    __shared__ float hid[128];
    __shared__ float red[256];
    int g = blockIdx.x, t = threadIdx.x;
    pr[t] = pooled[g * 256 + t];
    __syncthreads();
    if (t < 128) {
        float acc = b1[t];
        for (int i = 0; i < 256; ++i) acc += pr[i] * W1[i * 128 + t];
        hid[t] = fmaxf(acc, 0.f);
    }
    __syncthreads();
    float acc = b2[t];
    for (int i = 0; i < 128; ++i) acc += hid[i] * W2[i * 256 + t];
    red[t] = acc * acc;
    __syncthreads();
    for (int off = 128; off > 0; off >>= 1) {
        if (t < off) red[t] += red[t + off];
        __syncthreads();
    }
    float nrm = sqrtf(red[0]);
    out[g * 256 + t] = acc / fmaxf(nrm, 1e-12f);
}

// ---------------------------------------------------------------- launcher
extern "C" void kernel_launch(void* const* d_in, const int* in_sizes, int n_in,
                              void* d_out, int out_size, void* d_ws, size_t ws_size,
                              hipStream_t stream) {
    const float* x    = (const float*)d_in[0];
    const int* esrc   = (const int*)d_in[1];
    const int* edst   = (const int*)d_in[2];
    const int* gbatch = (const int*)d_in[3];
    const float *Wn[3], *Ws[3], *bb[3], *lg[3], *lb[3];
    for (int i = 0; i < 3; ++i) {
        Wn[i] = (const float*)d_in[4 + 5 * i];
        Ws[i] = (const float*)d_in[5 + 5 * i];
        bb[i] = (const float*)d_in[6 + 5 * i];
        lg[i] = (const float*)d_in[7 + 5 * i];
        lb[i] = (const float*)d_in[8 + 5 * i];
    }
    const float* W_ro1 = (const float*)d_in[19];
    const float* b_ro1 = (const float*)d_in[20];
    const float* W_ro2 = (const float*)d_in[21];
    const float* b_ro2 = (const float*)d_in[22];

    char* ws = (char*)d_ws;
    float* h0           = (float*)(ws);                       // 25,600,000 B
    float* h1           = (float*)(ws + 25600000);            // 25,600,000 B
    float* agg          = (float*)(ws + 51200000);            // 25,600,000 B
    float* cnt          = (float*)(ws + 76800000);            //    204,800 B (padded)
    unsigned short* wp0 = (unsigned short*)(ws + 77004800);   //     65,536 B
    unsigned short* wp1 = (unsigned short*)(ws + 77070336);
    unsigned short* wp2 = (unsigned short*)(ws + 77135872);
    float* gsum         = (float*)(ws + 77201408);            //    131,072 B
    unsigned* gmax      = (unsigned*)(ws + 77332480);         //    131,072 B
    float* gcnt         = (float*)(ws + 77463552);            //      1,024 B
    float* pooled       = (float*)(ws + 77464576);            //    262,144 B
    unsigned short* wp[3] = {wp0, wp1, wp2};

    // pack weights (B fragments) for the 3 layers
    for (int l = 0; l < 3; ++l)
        pack_w<<<64, 32, 0, stream>>>(Wn[l], Ws[l], wp[l]);

    const float* hin = x;
    float* houts[3] = {h0, h1, h0};
    for (int l = 0; l < 3; ++l) {
        // agg + cnt are contiguous: zero both in one pass
        zero_f<<<4096, 256, 0, stream>>>(agg, N_NODES * HID + 51200); // covers cnt pad
        scatter_edges<<<(N_EDGES + 7) / 8, 256, 0, stream>>>(hin, esrc, edst, agg, cnt);
        sage_gemm_ln_relu<<<(N_NODES + 15) / 16, 256, 0, stream>>>(
            agg, cnt, hin, wp[l], bb[l], lg[l], lb[l], houts[l]);
        hin = houts[l];
    }

    // zero gsum | gmax | gcnt (contiguous region)
    zero_f<<<1024, 256, 0, stream>>>(gsum, N_GRAPHS * HID * 2 + N_GRAPHS);
    pool_nodes<<<N_NODES, HID, 0, stream>>>(hin, gbatch, gsum, gmax, gcnt);
    pool_finalize<<<N_GRAPHS, HID, 0, stream>>>(gsum, gmax, gcnt, pooled);
    readout<<<N_GRAPHS, 256, 0, stream>>>(pooled, W_ro1, b_ro1, W_ro2, b_ro2,
                                          (float*)d_out);
}
